// MultiHeadAttention2D_73143293051615
// MI455X (gfx1250) — matrix operand order repaired and checked by reading, not compiled, not run on hardware
//
#include <hip/hip_runtime.h>

// ---------------------------------------------------------------------------
// MultiHeadAttention2D (cosine-distance attention) for gfx1250 / MI455X.
//   B=8, C=256, H=W=32 (N=1024), NH=8, DK=DV=32.
// Attention GEMMs use v_wmma_f32_16x16x32_f16 (K=32 == head dim for scores;
// chained K=1024 for out = P x V^T). q/k are unit-normalized before f16
// conversion (cosine attention is scale invariant, so /temp and norms fold).
// Scores in [0,1] -> softmax without max-shift, P stored in-place in 64KB LDS.
// ---------------------------------------------------------------------------

typedef __attribute__((ext_vector_type(16))) _Float16 v16h;
typedef __attribute__((ext_vector_type(8)))  _Float16 v8h;
typedef __attribute__((ext_vector_type(8)))  float    v8f;

#define B_   8
#define C_   256
#define N_   1024
#define NH_  8
#define DK_  32

__device__ __forceinline__ float mishf(float y) {
    // mish(y) = y * tanh(softplus(y)); softplus(y) ~= y for large y
    float sp = (y > 20.0f) ? y : log1pf(expf(y));
    return y * tanhf(sp);
}

__device__ __forceinline__ v16h cat16(v8h lo, v8h hi) {
    return __builtin_shufflevector(lo, hi, 0, 1, 2, 3, 4, 5, 6, 7,
                                           8, 9, 10, 11, 12, 13, 14, 15);
}

// ---------------------------------------------------------------------------
// Kernel 0: zero the attention-map accumulator
// ---------------------------------------------------------------------------
__global__ void zero_kernel(float* __restrict__ p, int n) {
    int i = blockIdx.x * 256 + threadIdx.x;
    if (i < n) p[i] = 0.0f;
}

// ---------------------------------------------------------------------------
// Kernel 1: grouped 1x1 convs -> mish -> (q,k normalize) -> f16 q̂,k̂,v^T
//   qhat/khat: [b][h][n][d]  (d contiguous, 32 halves/row)
//   vT:        [b][h][d][m]  (m contiguous)
// One block per (b, head g). 256 threads, 4 column-chunks of 256.
// ---------------------------------------------------------------------------
__global__ __launch_bounds__(256) void qkv_kernel(
    const float* __restrict__ x,
    const float* __restrict__ qw, const float* __restrict__ qb0,
    const float* __restrict__ qs0, const float* __restrict__ qbe0,
    const float* __restrict__ kw, const float* __restrict__ kb0,
    const float* __restrict__ ks0, const float* __restrict__ kbe0,
    const float* __restrict__ vw, const float* __restrict__ vb0,
    const float* __restrict__ vs0, const float* __restrict__ vbe0,
    _Float16* __restrict__ qhat, _Float16* __restrict__ khat,
    _Float16* __restrict__ vT)
{
    __shared__ float Wq[1024], Wk[1024], Wv[1024];
    __shared__ float xs[32 * 256];
    __shared__ float pq[96], pk[96], pv[96];  // [bias | scale | beta] x 32

    const int t = threadIdx.x;
    const int b = blockIdx.x >> 3;
    const int g = blockIdx.x & 7;

    for (int i = t; i < 1024; i += 256) {
        Wq[i] = qw[g * 1024 + i];
        Wk[i] = kw[g * 1024 + i];
        Wv[i] = vw[g * 1024 + i];
    }
    if (t < 32) {
        pq[t] = qb0[g * 32 + t]; pq[32 + t] = qs0[g * 32 + t]; pq[64 + t] = qbe0[g * 32 + t];
        pk[t] = kb0[g * 32 + t]; pk[32 + t] = ks0[g * 32 + t]; pk[64 + t] = kbe0[g * 32 + t];
        pv[t] = vb0[g * 32 + t]; pv[32 + t] = vs0[g * 32 + t]; pv[64 + t] = vbe0[g * 32 + t];
    }
    __syncthreads();

    for (int cc = 0; cc < 4; ++cc) {
        for (int idx = t; idx < 32 * 256; idx += 256) {
            int c = idx >> 8, col = idx & 255;
            xs[idx] = x[(size_t)(b * C_ + g * 32 + c) * N_ + cc * 256 + col];
        }
        __syncthreads();

        const int n = cc * 256 + t;
        float xc[32];
#pragma unroll
        for (int c = 0; c < 32; ++c) xc[c] = xs[c * 256 + t];

        // pass 1: norms of q,k rows + emit v
        float qn2 = 0.0f, kn2 = 0.0f;
        for (int d = 0; d < 32; ++d) {
            float aq = 0.0f, ak = 0.0f, av = 0.0f;
#pragma unroll
            for (int c = 0; c < 32; ++c) {
                aq = fmaf(Wq[d * 32 + c], xc[c], aq);
                ak = fmaf(Wk[d * 32 + c], xc[c], ak);
                av = fmaf(Wv[d * 32 + c], xc[c], av);
            }
            float yq = mishf(pq[32 + d] * (aq + pq[d]) + pq[64 + d]);
            float yk = mishf(pk[32 + d] * (ak + pk[d]) + pk[64 + d]);
            float yv = mishf(pv[32 + d] * (av + pv[d]) + pv[64 + d]);
            qn2 = fmaf(yq, yq, qn2);
            kn2 = fmaf(yk, yk, kn2);
            vT[(size_t)((b * NH_ + g) * 32 + d) * N_ + n] = (_Float16)yv;
        }
        // /temp on q cancels under cosine normalization -> skipped
        const float qi = rsqrtf(fmaxf(qn2, 1e-24f));
        const float ki = rsqrtf(fmaxf(kn2, 1e-24f));

        _Float16* qrow = qhat + ((size_t)(b * NH_ + g) * N_ + n) * 32;
        _Float16* krow = khat + ((size_t)(b * NH_ + g) * N_ + n) * 32;
        for (int d = 0; d < 32; ++d) {
            float aq = 0.0f, ak = 0.0f;
#pragma unroll
            for (int c = 0; c < 32; ++c) {
                aq = fmaf(Wq[d * 32 + c], xc[c], aq);
                ak = fmaf(Wk[d * 32 + c], xc[c], ak);
            }
            float yq = mishf(pq[32 + d] * (aq + pq[d]) + pq[64 + d]);
            float yk = mishf(pk[32 + d] * (ak + pk[d]) + pk[64 + d]);
            qrow[d] = (_Float16)(yq * qi);
            krow[d] = (_Float16)(yk * ki);
        }
        __syncthreads();
    }
}

// ---------------------------------------------------------------------------
// Kernel 2: fused attention per (b, h, i-block of 32 queries). 8 waves.
//   Phase 1: S[i][m] = 1 - acos(clip(q̂_i . k̂_m))/pi   (WMMA f16, K=32)
//   Phase 2: per-row softmax in-place in LDS (scores in [0,1], no max-shift)
//   Phase 3: waves 0-3: out^T = P x V^T (WMMA f16 chain, K=1024)
//            waves 4-7: attention-map column sums -> global atomics
// Dynamic LDS: 32 x 1024 f16 = 64 KB.
// ---------------------------------------------------------------------------
__global__ __launch_bounds__(256) void attn_kernel(
    const _Float16* __restrict__ qhat, const _Float16* __restrict__ khat,
    const _Float16* __restrict__ vT,
    float* __restrict__ outbuf, float* __restrict__ am_accum)
{
    extern __shared__ char smem_raw[];
    _Float16* S = (_Float16*)smem_raw;  // [32][1024]

    const int t     = threadIdx.x;
    const int lane  = t & 31;
    const int w     = t >> 5;
    const int bh    = blockIdx.x >> 5;     // b*NH + h
    const int i0    = (blockIdx.x & 31) * 32;
    const int b     = bh >> 3;
    const int h     = bh & 7;
    const int lhalf = (lane >= 16) ? 1 : 0;
    const int lm    = lane & 15;

    // ---- Phase 1: scores --------------------------------------------------
    {
        const int ti = w & 1;  // i-tile within block
        const _Float16* qrow = qhat + ((size_t)bh * N_ + i0 + ti * 16 + lm) * 32;
        v8h alo = *(const v8h*)(qrow + lhalf * 8);
        v8h ahi = *(const v8h*)(qrow + 16 + lhalf * 8);
        v16h afrag = cat16(alo, ahi);

        for (int st = 0; st < 16; ++st) {
            const int mt = (w >> 1) + 4 * st;  // m-tile 0..63
            const _Float16* krow =
                khat + ((size_t)bh * N_ + mt * 16 + lm) * 32;
            v8h blo = *(const v8h*)(krow + lhalf * 8);
            v8h bhi = *(const v8h*)(krow + 16 + lhalf * 8);
            v16h bfrag = cat16(blo, bhi);

            v8f c = {};
            c = __builtin_amdgcn_wmma_f32_16x16x32_f16(
                    false, afrag, false, bfrag, (short)0, c, false, false);
#pragma unroll
            for (int r = 0; r < 8; ++r) {
                float cosv = fminf(fmaxf(c[r], -0.999999f), 0.999999f);
                float sc = 1.0f - acosf(cosv) * 0.3183098861837907f;  // 1/pi
                int irow = ti * 16 + r + lhalf * 8;
                S[irow * N_ + mt * 16 + lm] = (_Float16)sc;
            }
        }
    }
    __syncthreads();

    // ---- Phase 2: softmax (rows owned exclusively per wave) ---------------
    for (int rr = 0; rr < 4; ++rr) {
        const int row = w * 4 + rr;
        _Float16* Sr = S + row * N_;
        float sum = 0.0f;
        for (int j = 0; j < 32; ++j) {
            int m = lane + 32 * j;
            float e = expf((float)Sr[m]);  // s in [0,1] -> no overflow risk
            Sr[m] = (_Float16)e;
            sum += e;
        }
#pragma unroll
        for (int off = 16; off >= 1; off >>= 1)
            sum += __shfl_xor(sum, off, 32);
        const float inv = 1.0f / sum;
        for (int j = 0; j < 32; ++j) {
            int m = lane + 32 * j;
            Sr[m] = (_Float16)((float)Sr[m] * inv);
        }
    }
    __syncthreads();

    // ---- Phase 3: out GEMM (waves 0-3) + attn-map sums (waves 4-7) --------
    if (w < 4) {
        const int oti = w & 1;   // i-tile (queries)
        const int otd = w >> 1;  // d-tile (values)
        v8f c = {};
        for (int st = 0; st < 32; ++st) {
            const int k0 = st * 32;
            const _Float16* Sr = S + (oti * 16 + lm) * N_ + k0;
            v8h alo = *(const v8h*)(Sr + lhalf * 8);
            v8h ahi = *(const v8h*)(Sr + 16 + lhalf * 8);
            v16h af = cat16(alo, ahi);

            const _Float16* vrow =
                vT + ((size_t)bh * 32 + otd * 16 + lm) * N_ + k0;
            v8h blo = *(const v8h*)(vrow + lhalf * 8);
            v8h bhi = *(const v8h*)(vrow + 16 + lhalf * 8);
            v16h bf = cat16(blo, bhi);

            c = __builtin_amdgcn_wmma_f32_16x16x32_f16(
                    false, af, false, bf, (short)0, c, false, false);
        }
        const int d = otd * 16 + lm;
#pragma unroll
        for (int r = 0; r < 8; ++r) {
            int irel = oti * 16 + r + lhalf * 8;
            outbuf[(size_t)(b * C_ + h * 32 + d) * N_ + i0 + irel] = c[r];
        }
    } else {
        const int ww = w - 4;
        for (int j = 0; j < 8; ++j) {
            const int m = ww * 256 + lane + 32 * j;
            float cs = 0.0f;
            for (int i = 0; i < 32; ++i) cs += (float)S[i * N_ + m];
            atomicAdd(&am_accum[b * N_ + m], cs);
        }
    }
}

// ---------------------------------------------------------------------------
// Kernel 3: final 1x1 conv (256x256) + affine + mish + residual -> y
// One block per (b, 32-column tile).
// ---------------------------------------------------------------------------
__global__ __launch_bounds__(256) void fconv_kernel(
    const float* __restrict__ outbuf, const float* __restrict__ fw,
    const float* __restrict__ fb, const float* __restrict__ fs,
    const float* __restrict__ fbe, const float* __restrict__ x,
    float* __restrict__ y)
{
    __shared__ float tile[256 * 32];
    const int t = threadIdx.x;
    const int b = blockIdx.x >> 5;
    const int ct = blockIdx.x & 31;

    for (int idx = t; idx < 256 * 32; idx += 256) {
        int c = idx >> 5, col = idx & 31;
        tile[idx] = outbuf[(size_t)(b * C_ + c) * N_ + ct * 32 + col];
    }
    __syncthreads();

    const int ch = t;
    float acc[32];
    const float bias = fb[ch];
#pragma unroll
    for (int col = 0; col < 32; ++col) acc[col] = bias;

    for (int c = 0; c < 256; ++c) {
        float wv = fw[ch * 256 + c];
#pragma unroll
        for (int col = 0; col < 32; ++col)
            acc[col] = fmaf(wv, tile[c * 32 + col], acc[col]);
    }
    const float sch = fs[ch], bch = fbe[ch];
#pragma unroll
    for (int col = 0; col < 32; ++col) {
        int n = ct * 32 + col;
        float yv = mishf(sch * acc[col] + bch);
        size_t o = (size_t)(b * C_ + ch) * N_ + n;
        y[o] = yv + x[o];
    }
}

// ---------------------------------------------------------------------------
// Kernel 4: min-max normalize the attention map per batch (scale-invariant
// to the /NH mean, so the head-mean factor is dropped).
// ---------------------------------------------------------------------------
__global__ __launch_bounds__(256) void amfin_kernel(
    const float* __restrict__ am, float* __restrict__ out)
{
    __shared__ float rmin[256], rmax[256];
    const int t = threadIdx.x, b = blockIdx.x;
    float v[4];
    float mn = 3.4e38f, mx = -3.4e38f;
#pragma unroll
    for (int j = 0; j < 4; ++j) {
        v[j] = am[b * N_ + t * 4 + j];
        mn = fminf(mn, v[j]);
        mx = fmaxf(mx, v[j]);
    }
    rmin[t] = mn; rmax[t] = mx;
    __syncthreads();
    for (int s = 128; s >= 1; s >>= 1) {
        if (t < s) {
            rmin[t] = fminf(rmin[t], rmin[t + s]);
            rmax[t] = fmaxf(rmax[t], rmax[t + s]);
        }
        __syncthreads();
    }
    const float lo = rmin[0];
    const float inv = 1.0f / (rmax[0] - rmin[0]);
#pragma unroll
    for (int j = 0; j < 4; ++j)
        out[b * N_ + t * 4 + j] = (v[j] - lo) * inv;
}

// ---------------------------------------------------------------------------
extern "C" void kernel_launch(void* const* d_in, const int* in_sizes, int n_in,
                              void* d_out, int out_size, void* d_ws, size_t ws_size,
                              hipStream_t stream)
{
    (void)in_sizes; (void)n_in; (void)out_size; (void)ws_size;

    const float* x   = (const float*)d_in[0];
    const float* qw  = (const float*)d_in[1];
    const float* qb  = (const float*)d_in[2];
    const float* qs  = (const float*)d_in[3];
    const float* qbe = (const float*)d_in[4];
    const float* kw  = (const float*)d_in[5];
    const float* kb  = (const float*)d_in[6];
    const float* ks  = (const float*)d_in[7];
    const float* kbe = (const float*)d_in[8];
    const float* vw  = (const float*)d_in[9];
    const float* vb  = (const float*)d_in[10];
    const float* vs  = (const float*)d_in[11];
    const float* vbe = (const float*)d_in[12];
    const float* fw  = (const float*)d_in[13];
    const float* fb  = (const float*)d_in[14];
    const float* fs  = (const float*)d_in[15];
    const float* fbe = (const float*)d_in[16];

    char* ws = (char*)d_ws;
    _Float16* qhat = (_Float16*)(ws);                       // 4 MB
    _Float16* khat = (_Float16*)(ws + ((size_t)4 << 20));   // 4 MB
    _Float16* vT   = (_Float16*)(ws + ((size_t)8 << 20));   // 4 MB
    float* outbuf  = (float*)(ws + ((size_t)12 << 20));     // 8 MB
    float* am_acc  = (float*)(ws + ((size_t)20 << 20));     // 32 KB

    float* y      = (float*)d_out;
    float* am_out = (float*)d_out + (size_t)B_ * C_ * N_;

    zero_kernel<<<(B_ * N_ + 255) / 256, 256, 0, stream>>>(am_acc, B_ * N_);

    qkv_kernel<<<B_ * NH_, 256, 0, stream>>>(
        x, qw, qb, qs, qbe, kw, kb, ks, kbe, vw, vb, vs, vbe, qhat, khat, vT);

    attn_kernel<<<B_ * NH_ * 32, 256, 65536, stream>>>(
        qhat, khat, vT, outbuf, am_acc);

    fconv_kernel<<<B_ * 32, 256, 0, stream>>>(outbuf, fw, fb, fs, fbe, x, y);

    amfin_kernel<<<B_, 256, 0, stream>>>(am_acc, am_out);
}
